// MultiheadAttention_62062277427459
// MI455X (gfx1250) — matrix operand order repaired and checked by reading, not compiled, not run on hardware
//
#include <hip/hip_runtime.h>
#include <hip/hip_bf16.h>

// ---------------------------------------------------------------------------
// MHA block for MI455X (gfx1250, wave32, WMMA).
//   B=2, S=2048, H=16, dh=64, D=1024, HB=H*B=32, M=B*S=4096
// ---------------------------------------------------------------------------

#define B_SZ    2
#define S_LEN   2048
#define NHEAD   16
#define HDIM    64
#define DMODEL  1024
#define HB      (NHEAD * B_SZ)      // 32
#define MROWS   (B_SZ * S_LEN)      // 4096
#define MASK_NUM (-4294967295.0f)   // -2^32 + 1

typedef __attribute__((ext_vector_type(2)))  __bf16 v2bf;
typedef __attribute__((ext_vector_type(4)))  __bf16 v4bf;
typedef __attribute__((ext_vector_type(8)))  __bf16 v8bf;
typedef __attribute__((ext_vector_type(16))) __bf16 v16bf;
typedef __attribute__((ext_vector_type(8)))  float  v8f;
typedef __attribute__((ext_vector_type(4)))  int    v4i;

__device__ __forceinline__ v16bf cat16(v8bf lo, v8bf hi) {
  return __builtin_shufflevector(lo, hi, 0,1,2,3,4,5,6,7,8,9,10,11,12,13,14,15);
}
__device__ __forceinline__ v8f wmma_bf16(v16bf a, v16bf b, v8f c) {
  // D = A(16x32 bf16) * B(32x16 bf16) + C(16x16 f32)
  return __builtin_amdgcn_wmma_f32_16x16x32_bf16(false, a, false, b, (short)0, c, false, false);
}
__device__ __forceinline__ float rowmax16(float x) {
  x = fmaxf(x, __shfl_xor(x, 1));
  x = fmaxf(x, __shfl_xor(x, 2));
  x = fmaxf(x, __shfl_xor(x, 4));
  x = fmaxf(x, __shfl_xor(x, 8));
  return x;
}
__device__ __forceinline__ float rowsum16(float x) {
  x += __shfl_xor(x, 1);
  x += __shfl_xor(x, 2);
  x += __shfl_xor(x, 4);
  x += __shfl_xor(x, 8);
  return x;
}

#if __has_builtin(__builtin_amdgcn_global_load_async_to_lds_b128)
#define HAVE_ASYNC_LDS 1
#else
#define HAVE_ASYNC_LDS 0
#endif

__device__ __forceinline__ void wait_asynccnt0() {
#if __has_builtin(__builtin_amdgcn_s_wait_asynccnt)
  __builtin_amdgcn_s_wait_asynccnt(0);
#else
  asm volatile("s_wait_asynccnt 0x0" ::: "memory");
#endif
}

#if HAVE_ASYNC_LDS
// async copy of 16 bytes global -> LDS (per lane)
__device__ __forceinline__ void async_copy_b128(const void* gsrc, void* ldst) {
  __builtin_amdgcn_global_load_async_to_lds_b128(
      (__attribute__((address_space(1))) v4i*)gsrc,
      (__attribute__((address_space(3))) v4i*)ldst, 0, 0);
}
#endif

// ---------------------------------------------------------------------------
// Kernel 1: projection GEMM  C[m,n] = A[m,:] * W[:,n]
// A: [MROWS, 1024] f32, W: [1024, 1024] f32.
// Output: bf16 in split-head layout out[((h*B + b)*S + s)*64 + dh].
// Block: 256 threads (8 waves), tile 128x128, K staged 64 at a time.
// 16 WMMAs per wave between each barrier pair.
// ---------------------------------------------------------------------------
__global__ __launch_bounds__(256) void gemm_proj(const float* __restrict__ A,
                                                 const float* __restrict__ W,
                                                 __bf16* __restrict__ out) {
  __shared__ alignas(16) __bf16 At[128][72];  // [m][k], row stride 144B (16B mult)
  __shared__ alignas(16) __bf16 Bt[128][72];  // [n][k] (W tile transposed)

  const int tid  = threadIdx.x;
  const int lane = tid & 31;
  const int wave = tid >> 5;
  const int half = lane >> 4;
  const int l16  = lane & 15;
  const int m0 = blockIdx.y * 128;
  const int n0 = blockIdx.x * 128;
  const int wrow = wave >> 1;     // 0..3 -> 32 rows each
  const int wcol = wave & 1;      // 0..1 -> 64 cols each

  v8f acc[2][4] = {};

  // A staging map: thread -> (row, 32-col half)
  const int rowA = tid >> 1;
  const int colA = (tid & 1) * 32;
  // W staging map: thread -> 4(k) x 8(n) sub-block
  const int nbW = (tid & 15) * 8;
  const int kbW = (tid >> 4) * 4;

  for (int k0 = 0; k0 < DMODEL; k0 += 64) {
    // ---- stage A tile (f32 -> bf16, packed b64 LDS stores) ----
    {
      const float* src = A + (size_t)(m0 + rowA) * DMODEL + k0 + colA;
      __builtin_prefetch(src + 64, 0, 1);
#pragma unroll
      for (int j = 0; j < 32; j += 4) {
        float4 f = *(const float4*)(src + j);
        v4bf p = { (__bf16)f.x, (__bf16)f.y, (__bf16)f.z, (__bf16)f.w };
        *(v4bf*)&At[rowA][colA + j] = p;
      }
    }
    // ---- stage W tile transposed (4k x 8n block per thread) ----
    {
      float wv[4][8];
#pragma unroll
      for (int kk = 0; kk < 4; kk++) {
        const float* src = W + (size_t)(k0 + kbW + kk) * DMODEL + n0 + nbW;
        __builtin_prefetch(src + (size_t)64 * DMODEL, 0, 1);
        float4 f0 = *(const float4*)(src);
        float4 f1 = *(const float4*)(src + 4);
        wv[kk][0] = f0.x; wv[kk][1] = f0.y; wv[kk][2] = f0.z; wv[kk][3] = f0.w;
        wv[kk][4] = f1.x; wv[kk][5] = f1.y; wv[kk][6] = f1.z; wv[kk][7] = f1.w;
      }
#pragma unroll
      for (int j = 0; j < 8; j++) {
        v4bf p = { (__bf16)wv[0][j], (__bf16)wv[1][j], (__bf16)wv[2][j], (__bf16)wv[3][j] };
        *(v4bf*)&Bt[nbW + j][kbW] = p;
      }
    }
    __syncthreads();

    // ---- 2 K-steps of 32 -> 16 WMMAs per wave per stage ----
#pragma unroll
    for (int ks = 0; ks < 2; ks++) {
      const int kb = ks * 32;
      v16bf afrag[2], bfrag[4];
#pragma unroll
      for (int mi = 0; mi < 2; mi++) {
        const int r = wrow * 32 + mi * 16 + l16;
        v8bf lo = *(const v8bf*)&At[r][kb + half * 8];
        v8bf hi = *(const v8bf*)&At[r][kb + 16 + half * 8];
        afrag[mi] = cat16(lo, hi);
      }
#pragma unroll
      for (int ni = 0; ni < 4; ni++) {
        const int n = wcol * 64 + ni * 16 + l16;
        v8bf lo = *(const v8bf*)&Bt[n][kb + half * 8];
        v8bf hi = *(const v8bf*)&Bt[n][kb + 16 + half * 8];
        bfrag[ni] = cat16(lo, hi);
      }
#pragma unroll
      for (int mi = 0; mi < 2; mi++)
#pragma unroll
        for (int ni = 0; ni < 4; ni++)
          acc[mi][ni] = wmma_bf16(afrag[mi], bfrag[ni], acc[mi][ni]);
    }
    __syncthreads();
  }

  // ---- store: bf16 split-head layout ----
#pragma unroll
  for (int mi = 0; mi < 2; mi++) {
#pragma unroll
    for (int ni = 0; ni < 4; ni++) {
      const int gcol = n0 + wcol * 64 + ni * 16 + l16;
      const int h  = gcol >> 6;
      const int dh = gcol & 63;
#pragma unroll
      for (int v = 0; v < 8; v++) {
        const int gr = m0 + wrow * 32 + mi * 16 + v + 8 * half;
        const int b  = gr >> 11;       // / S_LEN
        const int s  = gr & 2047;      // % S_LEN
        out[(((size_t)(h * B_SZ + b) * S_LEN + s) << 6) + dh] = (__bf16)acc[mi][ni][v];
      }
    }
  }
}

// ---------------------------------------------------------------------------
// Kernel 2: padding masks  qmask/kmask[hb*S + s] = sign(sum |proj|)
// ---------------------------------------------------------------------------
__global__ __launch_bounds__(256) void mask_kernel(const __bf16* __restrict__ q,
                                                   const __bf16* __restrict__ k,
                                                   float* __restrict__ qm,
                                                   float* __restrict__ km) {
  const int idx = blockIdx.x * 256 + threadIdx.x;    // 0 .. HB*S-1
  const v8bf* qp = (const v8bf*)(q + (size_t)idx * HDIM);
  const v8bf* kp = (const v8bf*)(k + (size_t)idx * HDIM);
  float sq = 0.f, sk = 0.f;
#pragma unroll
  for (int i = 0; i < 8; i++) {
    v8bf xq = qp[i], xk = kp[i];
#pragma unroll
    for (int j = 0; j < 8; j++) {
      sq += fabsf((float)xq[j]);
      sk += fabsf((float)xk[j]);
    }
  }
  qm[idx] = (sq > 0.f) ? 1.f : 0.f;
  km[idx] = (sk > 0.f) ? 1.f : 0.f;
}

// ---------------------------------------------------------------------------
// Kernel 3: flash attention per head-batch.
// Block: 128 threads = 4 waves; each wave owns 16 query rows.
// grid = (S/64, HB). KV processed in chunks of 32.
// K chunk staged to LDS via async global->LDS copy (ASYNCcnt path).
// ---------------------------------------------------------------------------
__global__ __launch_bounds__(128) void attn_kernel(const __bf16* __restrict__ qb,
                                                   const __bf16* __restrict__ kb,
                                                   const __bf16* __restrict__ vb,
                                                   const float* __restrict__ qmask,
                                                   const float* __restrict__ kmask,
                                                   float* __restrict__ att) {
  __shared__ alignas(16) __bf16 Kc[32][64];        // K chunk row-major [kv][dh]
  __shared__ alignas(16) __bf16 Vt[64][40];        // V chunk transposed [dh][kv+pad]
  __shared__ alignas(16) __bf16 Pt[4][16][40];     // per-wave P tile [row][kv+pad]

  const int tid  = threadIdx.x;
  const int lane = tid & 31;
  const int wave = tid >> 5;
  const int half = lane >> 4;
  const int l16  = lane & 15;
  const int hb   = blockIdx.y;
  const int q0   = blockIdx.x * 64 + wave * 16;

  // ---- load Q fragments once (A-layout, straight from global) ----
  const __bf16* qrow = qb + ((size_t)hb * S_LEN + q0 + l16) * HDIM;
  v16bf aq0 = cat16(*(const v8bf*)(qrow + half * 8),
                    *(const v8bf*)(qrow + 16 + half * 8));        // dh 0..31
  v16bf aq1 = cat16(*(const v8bf*)(qrow + 32 + half * 8),
                    *(const v8bf*)(qrow + 48 + half * 8));        // dh 32..63

  v8f o[4] = {};
  float mrow[8], lrow[8];
#pragma unroll
  for (int v = 0; v < 8; v++) { mrow[v] = -3.0e38f; lrow[v] = 0.f; }
  const float scale = 0.125f;   // 1/sqrt(64)

  const int eV = tid * 16;
  const int kvV = eV >> 6, dhV = eV & 63;

  for (int c = 0; c < S_LEN; c += 32) {
    // ---- stage K chunk (pure byte copy: async global->LDS when available) ----
    {
      const __bf16* gsrc = kb + ((size_t)hb * S_LEN + c) * HDIM + tid * 16;
      __bf16* ldst = ((__bf16*)Kc) + tid * 16;
#if HAVE_ASYNC_LDS
      async_copy_b128(gsrc, ldst);
      async_copy_b128(gsrc + 8, ldst + 8);
#else
      ((v8bf*)ldst)[0] = ((const v8bf*)gsrc)[0];
      ((v8bf*)ldst)[1] = ((const v8bf*)gsrc)[1];
#endif
    }
    // ---- stage V chunk transposed into LDS ----
    {
      const __bf16* src = vb + ((size_t)hb * S_LEN + c + kvV) * HDIM + dhV;
      v8bf r0 = *(const v8bf*)(src);
      v8bf r1 = *(const v8bf*)(src + 8);
#pragma unroll
      for (int j = 0; j < 8; j++) {
        Vt[dhV + j][kvV]     = r0[j];
        Vt[dhV + 8 + j][kvV] = r1[j];
      }
    }
#if HAVE_ASYNC_LDS
    wait_asynccnt0();
#endif
    __syncthreads();

    // ---- scores for both 16-wide tiles of this chunk (Q.K^T) ----
    v8f sc[2];
    float km[2];
#pragma unroll
    for (int t = 0; t < 2; t++) {
      const __bf16* krow = &Kc[t * 16 + l16][0];
      v16bf bk0 = cat16(*(const v8bf*)(krow + half * 8),
                        *(const v8bf*)(krow + 16 + half * 8));       // dh 0..31
      v16bf bk1 = cat16(*(const v8bf*)(krow + 32 + half * 8),
                        *(const v8bf*)(krow + 48 + half * 8));       // dh 32..63
      v8f s = {};
      s = wmma_bf16(aq0, bk0, s);
      s = wmma_bf16(aq1, bk1, s);
      sc[t] = s;
      km[t] = kmask[(size_t)hb * S_LEN + c + t * 16 + l16];
    }

    // ---- online softmax update per owned row ----
#pragma unroll
    for (int v = 0; v < 8; v++) {
      float s0 = (km[0] == 0.f) ? MASK_NUM : sc[0][v] * scale;
      float s1 = (km[1] == 0.f) ? MASK_NUM : sc[1][v] * scale;
      float mx = rowmax16(fmaxf(s0, s1));
      float mnew = fmaxf(mrow[v], mx);
      float corr = __expf(mrow[v] - mnew);
      float p0 = __expf(s0 - mnew);
      float p1 = __expf(s1 - mnew);
      float rs = rowsum16(p0 + p1);
      lrow[v] = lrow[v] * corr + rs;
      mrow[v] = mnew;
      o[0][v] *= corr; o[1][v] *= corr; o[2][v] *= corr; o[3][v] *= corr;
      Pt[wave][v + 8 * half][l16]      = (__bf16)p0;
      Pt[wave][v + 8 * half][16 + l16] = (__bf16)p1;
    }
    __syncthreads();

    // ---- O += P(16x32) @ V(32x64) ----
    {
      const __bf16* prow = &Pt[wave][l16][0];
      v16bf ap = cat16(*(const v8bf*)(prow + half * 8),
                       *(const v8bf*)(prow + 16 + half * 8));
#pragma unroll
      for (int d = 0; d < 4; d++) {
        const __bf16* vrow = &Vt[d * 16 + l16][0];
        v16bf bv = cat16(*(const v8bf*)(vrow + half * 8),
                         *(const v8bf*)(vrow + 16 + half * 8));
        o[d] = wmma_bf16(ap, bv, o[d]);
      }
    }
    __syncthreads();
  }

  // ---- finalize: qmask / l, store to merged [B, S, D] ----
  const int h = hb >> 1;      // hb = h*B + b, B=2
  const int b = hb & 1;
#pragma unroll
  for (int v = 0; v < 8; v++) {
    const int row = q0 + v + 8 * half;
    const float inv = qmask[(size_t)hb * S_LEN + row] / lrow[v];
#pragma unroll
    for (int d = 0; d < 4; d++) {
      att[((size_t)(b * S_LEN + row)) * DMODEL + h * HDIM + d * 16 + l16] = o[d][v] * inv;
    }
  }
}

// ---------------------------------------------------------------------------
// Kernel 4: residual + LayerNorm. One block (256 thr) per row of 1024.
// ---------------------------------------------------------------------------
__global__ __launch_bounds__(256) void ln_kernel(const float* __restrict__ att,
                                                 const float* __restrict__ resid,
                                                 const float* __restrict__ gamma,
                                                 const float* __restrict__ beta,
                                                 float* __restrict__ out) {
  __shared__ float red0[8], red1[8];
  const int row = blockIdx.x;
  const int tid = threadIdx.x;
  const float* a = att   + (size_t)row * DMODEL;
  const float* r = resid + (size_t)row * DMODEL;

  float x[4], s = 0.f, s2 = 0.f;
#pragma unroll
  for (int j = 0; j < 4; j++) {
    const int c = tid + j * 256;
    x[j] = a[c] + r[c];
    s += x[j]; s2 += x[j] * x[j];
  }
#pragma unroll
  for (int off = 1; off < 32; off <<= 1) {
    s  += __shfl_xor(s, off);
    s2 += __shfl_xor(s2, off);
  }
  if ((tid & 31) == 0) { red0[tid >> 5] = s; red1[tid >> 5] = s2; }
  __syncthreads();
  float ts = 0.f, ts2 = 0.f;
#pragma unroll
  for (int w = 0; w < 8; w++) { ts += red0[w]; ts2 += red1[w]; }
  const float mean = ts * (1.f / DMODEL);
  const float var  = ts2 * (1.f / DMODEL) - mean * mean;
  const float rstd = rsqrtf(var + 1e-8f);
#pragma unroll
  for (int j = 0; j < 4; j++) {
    const int c = tid + j * 256;
    out[(size_t)row * DMODEL + c] = gamma[c] * ((x[j] - mean) * rstd) + beta[c];
  }
}

// ---------------------------------------------------------------------------
// launch
// ---------------------------------------------------------------------------
extern "C" void kernel_launch(void* const* d_in, const int* in_sizes, int n_in,
                              void* d_out, int out_size, void* d_ws, size_t ws_size,
                              hipStream_t stream) {
  const float* queries = (const float*)d_in[0];
  const float* keys    = (const float*)d_in[1];
  const float* values  = (const float*)d_in[2];
  const float* Wq      = (const float*)d_in[3];
  const float* Wk      = (const float*)d_in[4];
  const float* Wv      = (const float*)d_in[5];
  const float* gamma   = (const float*)d_in[6];
  const float* beta    = (const float*)d_in[7];
  float* out = (float*)d_out;

  char* ws = (char*)d_ws;
  const size_t QKV_BYTES  = (size_t)HB * S_LEN * HDIM * sizeof(__bf16);   // 8 MB each
  const size_t MASK_BYTES = (size_t)HB * S_LEN * sizeof(float);           // 256 KB each
  __bf16* qb    = (__bf16*)(ws);
  __bf16* kb    = (__bf16*)(ws + QKV_BYTES);
  __bf16* vb    = (__bf16*)(ws + 2 * QKV_BYTES);
  float*  qmask = (float*)(ws + 3 * QKV_BYTES);
  float*  kmask = (float*)(ws + 3 * QKV_BYTES + MASK_BYTES);
  float*  att   = (float*)(ws + 3 * QKV_BYTES + 2 * MASK_BYTES);          // 16 MB f32

  dim3 ggrid(DMODEL / 128, MROWS / 128);
  gemm_proj<<<ggrid, 256, 0, stream>>>(queries, Wq, qb);
  gemm_proj<<<ggrid, 256, 0, stream>>>(keys,    Wk, kb);
  gemm_proj<<<ggrid, 256, 0, stream>>>(values,  Wv, vb);

  mask_kernel<<<(HB * S_LEN) / 256, 256, 0, stream>>>(qb, kb, qmask, kmask);

  attn_kernel<<<dim3(S_LEN / 64, HB), 128, 0, stream>>>(qb, kb, vb, qmask, kmask, att);

  ln_kernel<<<MROWS, 256, 0, stream>>>(att, queries, gamma, beta, out);
}
